// PGCoverageDecoder_59742995087549
// MI455X (gfx1250) — compile-verified, matrix operand order checked by reading, not run on hardware
//
#include <hip/hip_runtime.h>

#define Bz 32
#define Sz 400
#define Hz 512
#define Ez 256
#define Vz 50000
#define Tz 64
#define VEXTz 50050
#define NTILES 3125   // Vz / 16

typedef __attribute__((ext_vector_type(16))) __bf16 v16bf;
typedef __attribute__((ext_vector_type(8)))  float  v8f;
typedef unsigned uint4v __attribute__((ext_vector_type(4)));

__device__ __forceinline__ unsigned short f2bf_bits(float f) {
    unsigned u = __builtin_bit_cast(unsigned, f);
    u += 0x7FFFu + ((u >> 16) & 1u);          // round-to-nearest-even
    return (unsigned short)(u >> 16);
}
__device__ __forceinline__ float sigm(float x) { return 1.f / (1.f + __expf(-x)); }

// ---------------- prep kernels ----------------

__global__ void init_kernel(const float* __restrict__ hidden, const float* __restrict__ cell,
                            float* __restrict__ h, float* __restrict__ c,
                            float* __restrict__ cov, float* __restrict__ covacc) {
    int i = blockIdx.x * 256 + threadIdx.x;
    if (i < Bz * Hz) { h[i] = hidden[i]; c[i] = cell[i]; }
    if (i < Bz * Sz) cov[i] = 0.f;
    if (i == 0) covacc[0] = 0.f;
}

__global__ void convert_kernel(const float* __restrict__ W, unsigned short* __restrict__ Wb, long n) {
    long i = (long)blockIdx.x * 256 + threadIdx.x;
    if (i < n) Wb[i] = f2bf_bits(W[i]);
}

// enc_proj[b,s,k] = sum_x enc_out[b,s,x] * W_enc[x,k]
__global__ void encproj_kernel(const float* __restrict__ enc_out, const float* __restrict__ W_enc,
                               float* __restrict__ enc_proj) {
    size_t idx = (size_t)blockIdx.x * 256 + threadIdx.x;   // over B*S*H
    if (idx >= (size_t)Bz * Sz * Hz) return;
    int k = (int)(idx & (Hz - 1));
    size_t bs = idx >> 9;
    const float* e = enc_out + bs * Hz;
    float acc = 0.f;
    for (int x = 0; x < Hz; ++x) acc += e[x] * W_enc[x * Hz + k];
    enc_proj[idx] = acc;
}

// ---------------- per-step kernels ----------------

// one block per batch element
__global__ void attention_kernel(const float* __restrict__ enc_proj, const float* __restrict__ enc_out,
                                 const float* __restrict__ h, const float* __restrict__ W_dec,
                                 const float* __restrict__ w_cov, const float* __restrict__ v_att,
                                 const float* __restrict__ b_att, const int* __restrict__ src_lens,
                                 float* __restrict__ cov, float* __restrict__ attn,
                                 float* __restrict__ context, float* __restrict__ closs) {
    int b = blockIdx.x, tid = threadIdx.x;                // 256 threads
    __shared__ float sh[Hz], sdecW[Hz], svatt[Hz], swcov[Hz], sbatt[Hz];
    __shared__ float sscore[Sz];
    __shared__ float red[256];

    for (int k = tid; k < Hz; k += 256) {
        sh[k] = h[b * Hz + k]; svatt[k] = v_att[k];
        swcov[k] = w_cov[k];   sbatt[k] = b_att[k];
    }
    __syncthreads();
    for (int k = tid; k < Hz; k += 256) {                 // decW = h @ W_dec
        float acc = 0.f;
        for (int x = 0; x < Hz; ++x) acc += sh[x] * W_dec[x * Hz + k];
        sdecW[k] = acc;
    }
    __syncthreads();
    int len = src_lens[b];
    for (int s = tid; s < Sz; s += 256) {
        float cv = cov[b * Sz + s];
        const float* ep = enc_proj + ((size_t)b * Sz + s) * Hz;
        float acc = 0.f;
        for (int k = 0; k < Hz; ++k)
            acc += svatt[k] * tanhf(ep[k] + sdecW[k] + cv * swcov[k] + sbatt[k]);
        sscore[s] = (s < len) ? acc : -1e9f;
    }
    __syncthreads();
    // softmax over S
    float m = -3.4e38f;
    for (int s = tid; s < Sz; s += 256) m = fmaxf(m, sscore[s]);
    red[tid] = m; __syncthreads();
    for (int r = 128; r > 0; r >>= 1) { if (tid < r) red[tid] = fmaxf(red[tid], red[tid + r]); __syncthreads(); }
    m = red[0]; __syncthreads();
    float sum = 0.f;
    for (int s = tid; s < Sz; s += 256) { float e = __expf(sscore[s] - m); sscore[s] = e; sum += e; }
    red[tid] = sum; __syncthreads();
    for (int r = 128; r > 0; r >>= 1) { if (tid < r) red[tid] += red[tid + r]; __syncthreads(); }
    float inv = 1.f / red[0];
    __syncthreads();
    float cl = 0.f;
    for (int s = tid; s < Sz; s += 256) {
        float a = sscore[s] * inv;
        sscore[s] = a;
        attn[b * Sz + s] = a;
        float cv = cov[b * Sz + s];
        cl += fminf(a, cv);                                // coverage loss uses OLD cov
        cov[b * Sz + s] = cv + a;
    }
    red[tid] = cl; __syncthreads();
    for (int r = 128; r > 0; r >>= 1) { if (tid < r) red[tid] += red[tid + r]; __syncthreads(); }
    if (tid == 0) closs[b] = red[0];
    __syncthreads();
    // context = attn @ enc_out
    for (int k = tid; k < Hz; k += 256) {
        float acc = 0.f;
        for (int s = 0; s < Sz; ++s) acc += sscore[s] * enc_out[((size_t)b * Sz + s) * Hz + k];
        context[b * Hz + k] = acc;
    }
}

// gates[b,j] = [x|ctx] . W_ih[j,:] + b_ih[j] + h . W_hh[j,:] + b_hh[j]
__global__ void gates_kernel(const int* __restrict__ tgt, const float* __restrict__ emb,
                             const float* __restrict__ context, const float* __restrict__ h,
                             const float* __restrict__ W_ih, const float* __restrict__ W_hh,
                             const float* __restrict__ b_ih, const float* __restrict__ b_hh,
                             float* __restrict__ gates, int t) {
    int idx = blockIdx.x * 256 + threadIdx.x;
    if (idx >= Bz * 4 * Hz) return;
    int b = idx >> 11, j = idx & 2047;
    const float* xrow = emb + (size_t)tgt[b * Tz + t] * Ez;
    const float* ctx = context + b * Hz;
    const float* hrow = h + b * Hz;
    const float* wi = W_ih + (size_t)j * (Ez + Hz);
    const float* wh = W_hh + (size_t)j * Hz;
    float acc = b_ih[j] + b_hh[j];
    for (int k = 0; k < Ez; ++k) acc += xrow[k] * wi[k];
    for (int k = 0; k < Hz; ++k) acc += ctx[k] * wi[Ez + k];
    for (int k = 0; k < Hz; ++k) acc += hrow[k] * wh[k];
    gates[idx] = acc;
}

// one block (512 threads) per batch element: LSTM pointwise + pgen + deterministic cov-loss accum
__global__ void pointwise_kernel(const float* __restrict__ gates, float* __restrict__ h,
                                 float* __restrict__ c, const float* __restrict__ context,
                                 float* __restrict__ hctx, const int* __restrict__ tgt,
                                 const float* __restrict__ emb, const float* __restrict__ w_ctx,
                                 const float* __restrict__ w_dec_p, const float* __restrict__ w_emb_p,
                                 const float* __restrict__ b_pgen, float* __restrict__ pgen,
                                 const float* __restrict__ closs, float* __restrict__ covacc, int t) {
    int b = blockIdx.x, k = threadIdx.x;
    const float* g = gates + b * 4 * Hz;
    float ig = sigm(g[k]), fg = sigm(g[Hz + k]);
    float gg = tanhf(g[2 * Hz + k]), og = sigm(g[3 * Hz + k]);
    float c_new = fg * c[b * Hz + k] + ig * gg;
    float h_new = og * tanhf(c_new);
    c[b * Hz + k] = c_new;
    h[b * Hz + k] = h_new;
    float ctxv = context[b * Hz + k];
    hctx[b * 2 * Hz + k] = h_new;
    hctx[b * 2 * Hz + Hz + k] = ctxv;
    float part = ctxv * w_ctx[k] + h_new * w_dec_p[k];
    if (k < Ez) part += emb[(size_t)tgt[b * Tz + t] * Ez + k] * w_emb_p[k];
    __shared__ float red[Hz];
    red[k] = part; __syncthreads();
    for (int r = 256; r > 0; r >>= 1) { if (k < r) red[k] += red[k + r]; __syncthreads(); }
    if (k == 0) pgen[b] = sigm(red[0] + b_pgen[0]);
    if (b == 0 && k == 0) {                   // single-thread, deterministic accumulation
        float s = 0.f;
        for (int i = 0; i < Bz; ++i) s += closs[i];
        covacc[0] += s * (1.f / Bz);
    }
}

// logits[b,v] = hctx[b,:] . W_out[v,:] + b_out[v]  via bf16 WMMA; W_out bf16 streamed from L2
__global__ void __launch_bounds__(256) vocab_gemm_kernel(const float* __restrict__ hctx,
                                                         const unsigned short* __restrict__ Wb,
                                                         const float* __restrict__ b_out,
                                                         float* __restrict__ logits) {
    __shared__ unsigned short sA[Bz * 2 * Hz];            // 32x1024 bf16 = 64 KB LDS
    for (int i = threadIdx.x; i < Bz * 2 * Hz; i += 256) sA[i] = f2bf_bits(hctx[i]);
    __syncthreads();

    int wave = threadIdx.x >> 5;
    int lane = threadIdx.x & 31;
    int ntile = blockIdx.x * 8 + wave;
    if (ntile < NTILES) {
        int vb = ntile << 4;
        int n = lane & 15;
        int kh = (lane >> 4) << 3;                        // 0 or 8
        v8f acc0 = {}; v8f acc1 = {};
        const unsigned short* brow = Wb + (size_t)(vb + n) * (2 * Hz);
        union BF { v16bf v; uint4v q[2]; };
        #pragma unroll 4
        for (int k0 = 0; k0 < 2 * Hz; k0 += 32) {
            BF bf;                                         // B fragment: column vb+n
            bf.q[0] = *(const uint4v*)(brow + k0 + kh);
            bf.q[1] = *(const uint4v*)(brow + k0 + 16 + kh);
            BF a0, a1;                                     // A fragments from LDS
            const unsigned short* ap0 = sA + (size_t)n * (2 * Hz) + k0 + kh;
            const unsigned short* ap1 = sA + (size_t)(n + 16) * (2 * Hz) + k0 + kh;
            a0.q[0] = *(const uint4v*)(ap0);
            a0.q[1] = *(const uint4v*)(ap0 + 16);
            a1.q[0] = *(const uint4v*)(ap1);
            a1.q[1] = *(const uint4v*)(ap1 + 16);
            acc0 = __builtin_amdgcn_wmma_f32_16x16x32_bf16(false, a0.v, false, bf.v,
                                                           (short)0, acc0, false, false);
            acc1 = __builtin_amdgcn_wmma_f32_16x16x32_bf16(false, a1.v, false, bf.v,
                                                           (short)0, acc1, false, false);
        }
        float bias = b_out[vb + n];
        int rbase = (lane >> 4) * 8;                       // C layout: VGPR j -> M=j / M=8+j
        #pragma unroll
        for (int j = 0; j < 8; ++j) {
            int r = rbase + j;
            logits[(size_t)r * Vz + vb + n]        = acc0[j] + bias;
            logits[(size_t)(16 + r) * Vz + vb + n] = acc1[j] + bias;
        }
    }
}

// softmax over V, scale by pgen, write extended dist (+ zero OOV tail)
__global__ void finalize_kernel(const float* __restrict__ logits, const float* __restrict__ pgen,
                                float* __restrict__ out, int t) {
    int b = blockIdx.x, tid = threadIdx.x;                // 256 threads
    const float* lg = logits + (size_t)b * Vz;
    float* dist = out + ((size_t)b * (Tz - 1) + t) * VEXTz;
    __shared__ float red[256];
    float m = -3.4e38f;
    for (int v = tid; v < Vz; v += 256) m = fmaxf(m, lg[v]);
    red[tid] = m; __syncthreads();
    for (int r = 128; r > 0; r >>= 1) { if (tid < r) red[tid] = fmaxf(red[tid], red[tid + r]); __syncthreads(); }
    m = red[0]; __syncthreads();
    float sum = 0.f;
    for (int v = tid; v < Vz; v += 256) sum += __expf(lg[v] - m);
    red[tid] = sum; __syncthreads();
    for (int r = 128; r > 0; r >>= 1) { if (tid < r) red[tid] += red[tid + r]; __syncthreads(); }
    float scale = pgen[b] / red[0];
    for (int v = tid; v < Vz; v += 256) dist[v] = __expf(lg[v] - m) * scale;
    for (int v = Vz + tid; v < VEXTz; v += 256) dist[v] = 0.f;
}

// one lane per batch row, serial over S: deterministic even with duplicate OOV indices
__global__ void scatter_kernel(const int* __restrict__ oov, const float* __restrict__ attn,
                               const float* __restrict__ pgen, float* __restrict__ out, int t) {
    int b = threadIdx.x;
    if (b >= Bz) return;
    float* dist = out + ((size_t)b * (Tz - 1) + t) * VEXTz;
    float w = 1.f - pgen[b];
    for (int s = 0; s < Sz; ++s)
        dist[oov[b * Sz + s]] += w * attn[b * Sz + s];
}

__global__ void tail_kernel(const float* __restrict__ h, const float* __restrict__ c,
                            const float* __restrict__ covacc, float* __restrict__ out) {
    size_t base = (size_t)Bz * (Tz - 1) * VEXTz;
    int i = blockIdx.x * 256 + threadIdx.x;
    if (i < Bz * Hz) { out[base + i] = h[i]; out[base + Bz * Hz + i] = c[i]; }
    if (i == 0) out[base + 2 * Bz * Hz] = covacc[0] * (1.0f / (Tz - 1));   // COV_W = 1
}

// ---------------- launch ----------------

extern "C" void kernel_launch(void* const* d_in, const int* in_sizes, int n_in,
                              void* d_out, int out_size, void* d_ws, size_t ws_size,
                              hipStream_t stream) {
    const int*   tgt      = (const int*)d_in[0];
    const float* hidden   = (const float*)d_in[1];
    const float* cell     = (const float*)d_in[2];
    const float* enc_out  = (const float*)d_in[3];
    const int*   src_lens = (const int*)d_in[4];
    const int*   oov      = (const int*)d_in[5];
    const float* emb      = (const float*)d_in[6];
    const float* W_enc    = (const float*)d_in[7];
    const float* W_dec    = (const float*)d_in[8];
    const float* w_cov    = (const float*)d_in[9];
    const float* v_att    = (const float*)d_in[10];
    const float* b_att    = (const float*)d_in[11];
    const float* W_ih     = (const float*)d_in[12];
    const float* W_hh     = (const float*)d_in[13];
    const float* b_ih     = (const float*)d_in[14];
    const float* b_hh     = (const float*)d_in[15];
    const float* W_out    = (const float*)d_in[16];
    const float* b_out    = (const float*)d_in[17];
    const float* w_ctx    = (const float*)d_in[18];
    const float* w_dec_p  = (const float*)d_in[19];
    const float* w_emb_p  = (const float*)d_in[20];
    const float* b_pgen   = (const float*)d_in[21];
    float* out = (float*)d_out;
    (void)in_sizes; (void)n_in; (void)out_size; (void)ws_size;

    char* ws = (char*)d_ws;
    size_t off = 0;
    auto alloc = [&](size_t bytes) -> void* {
        void* p = ws + off;
        off += (bytes + 255) & ~(size_t)255;
        return p;
    };
    unsigned short* Wb  = (unsigned short*)alloc((size_t)Vz * 2 * Hz * 2); // 102.4 MB (fits L2)
    float* enc_proj = (float*)alloc((size_t)Bz * Sz * Hz * 4);             // 26.2 MB
    float* h        = (float*)alloc((size_t)Bz * Hz * 4);
    float* c        = (float*)alloc((size_t)Bz * Hz * 4);
    float* cov      = (float*)alloc((size_t)Bz * Sz * 4);
    float* attn     = (float*)alloc((size_t)Bz * Sz * 4);
    float* context  = (float*)alloc((size_t)Bz * Hz * 4);
    float* hctx     = (float*)alloc((size_t)Bz * 2 * Hz * 4);
    float* gates    = (float*)alloc((size_t)Bz * 4 * Hz * 4);
    float* logits   = (float*)alloc((size_t)Bz * Vz * 4);
    float* pgen     = (float*)alloc((size_t)Bz * 4);
    float* closs    = (float*)alloc((size_t)Bz * 4);
    float* covacc   = (float*)alloc(256);

    init_kernel<<<64, 256, 0, stream>>>(hidden, cell, h, c, cov, covacc);
    convert_kernel<<<((long)Vz * 2 * Hz + 255) / 256, 256, 0, stream>>>(W_out, Wb, (long)Vz * 2 * Hz);
    encproj_kernel<<<(Bz * Sz * Hz) / 256, 256, 0, stream>>>(enc_out, W_enc, enc_proj);

    for (int t = 0; t < Tz - 1; ++t) {
        attention_kernel<<<Bz, 256, 0, stream>>>(enc_proj, enc_out, h, W_dec, w_cov, v_att,
                                                 b_att, src_lens, cov, attn, context, closs);
        gates_kernel<<<(Bz * 4 * Hz) / 256, 256, 0, stream>>>(tgt, emb, context, h, W_ih, W_hh,
                                                              b_ih, b_hh, gates, t);
        pointwise_kernel<<<Bz, Hz, 0, stream>>>(gates, h, c, context, hctx, tgt, emb, w_ctx,
                                                w_dec_p, w_emb_p, b_pgen, pgen, closs, covacc, t);
        vocab_gemm_kernel<<<(NTILES + 7) / 8, 256, 0, stream>>>(hctx, Wb, b_out, logits);
        finalize_kernel<<<Bz, 256, 0, stream>>>(logits, pgen, out, t);
        scatter_kernel<<<1, 64, 0, stream>>>(oov, attn, pgen, out, t);
    }
    tail_kernel<<<64, 256, 0, stream>>>(h, c, covacc, out);
}
